// PointNet2Backbone_83425444757830
// MI455X (gfx1250) — compile-verified, matrix operand order checked by reading, not run on hardware
//
#include <hip/hip_runtime.h>
#include <hip/hip_bf16.h>
#include <cstdint>
#include <cstddef>

typedef __attribute__((ext_vector_type(16))) _Float16 v16h;
typedef __attribute__((ext_vector_type(8)))  _Float16 v8h;
typedef __attribute__((ext_vector_type(8)))  float    v8f;
typedef __attribute__((ext_vector_type(4)))  unsigned int v4u;
typedef __attribute__((ext_vector_type(8)))  int      v8i;
typedef __attribute__((ext_vector_type(4)))  int      v4i;

#define BN_EPS 1e-5f
#define KCH 1024   // K-chunk staged in LDS (16 rows x 1024 halfs = 32 KB)

static inline int rup(int x, int m) { return (x + m - 1) / m * m; }

// ---------------------------------------------------------------------------
// Farthest point sampling: one block (512 threads) per batch.
// ---------------------------------------------------------------------------
__global__ void fps_kernel(const float* __restrict__ xyz, float* __restrict__ dist,
                           int N, int npoint, int* __restrict__ out)
{
    int b = blockIdx.x;
    const float* px = xyz + (size_t)b * N * 3;
    float* pd = dist + (size_t)b * N;
    __shared__ float sd[512];
    __shared__ int   si[512];
    __shared__ int   sfar;
    int tid = threadIdx.x;
    int far = 0;
    for (int it = 0; it < npoint; ++it) {
        if (tid == 0) out[b * npoint + it] = far;
        float cx = px[far * 3 + 0], cy = px[far * 3 + 1], cz = px[far * 3 + 2];
        float bmax = -1.0f; int bidx = 0;
        for (int i = tid; i < N; i += blockDim.x) {
            float dx = px[i * 3 + 0] - cx;
            float dy = px[i * 3 + 1] - cy;
            float dz = px[i * 3 + 2] - cz;
            float d = dx * dx + dy * dy + dz * dz;
            float prev = (it == 0) ? 1e10f : pd[i];
            float dm = prev < d ? prev : d;
            pd[i] = dm;
            if (dm > bmax) { bmax = dm; bidx = i; }
        }
        sd[tid] = bmax; si[tid] = bidx;
        __syncthreads();
        for (int o = blockDim.x >> 1; o > 0; o >>= 1) {
            if (tid < o) {
                if (sd[tid + o] > sd[tid] ||
                    (sd[tid + o] == sd[tid] && si[tid + o] < si[tid])) {
                    sd[tid] = sd[tid + o]; si[tid] = si[tid + o];
                }
            }
            __syncthreads();
        }
        if (tid == 0) sfar = si[0];
        __syncthreads();
        far = sfar;
    }
}

__global__ void gather_xyz(const float* __restrict__ xyz, const int* __restrict__ idx,
                           int N, int S, int total, float* __restrict__ nx)
{
    int t = blockIdx.x * blockDim.x + threadIdx.x;
    if (t >= total) return;
    int b = t / S, s = t % S;
    int i = idx[b * S + s];
    nx[(size_t)t * 3 + 0] = xyz[((size_t)b * N + i) * 3 + 0];
    nx[(size_t)t * 3 + 1] = xyz[((size_t)b * N + i) * 3 + 1];
    nx[(size_t)t * 3 + 2] = xyz[((size_t)b * N + i) * 3 + 2];
}

// ---------------------------------------------------------------------------
// Ball query: one wave32 per center, ballot-ordered append.
// ---------------------------------------------------------------------------
__global__ void ball_query(const float* __restrict__ xyz, const float* __restrict__ nxyz,
                           int N, int S, int total, float r2, int nsample,
                           int* __restrict__ gi)
{
    int wave = blockIdx.x * (blockDim.x >> 5) + (threadIdx.x >> 5);
    if (wave >= total) return;
    int lane = threadIdx.x & 31;
    int b = wave / S;
    const float* px = xyz + (size_t)b * N * 3;
    const float* q = nxyz + (size_t)wave * 3;
    float qx = q[0], qy = q[1], qz = q[2];
    int* out = gi + (size_t)wave * nsample;
    int cnt = 0;
    for (int base = 0; base < N && cnt < nsample; base += 32) {
        int i = base + lane;
        bool ok = false;
        if (i < N) {
            float dx = px[i * 3 + 0] - qx;
            float dy = px[i * 3 + 1] - qy;
            float dz = px[i * 3 + 2] - qz;
            ok = (dx * dx + dy * dy + dz * dz <= r2);
        }
        unsigned m32 = __builtin_amdgcn_ballot_w32(ok);
        int pos = cnt + __popc(m32 & ((1u << lane) - 1u));
        if (ok && pos < nsample) out[pos] = i;
        cnt += __popc(m32);
    }
    if (cnt > nsample) cnt = nsample;
    int firstIdx = (cnt > 0) ? out[0] : 0;
    for (int p = cnt + lane; p < nsample; p += 32) out[p] = firstIdx;
}

// ---------------------------------------------------------------------------
// Build grouped features h = [points[gi], xyz[gi]-center] as f16 rows [Kpad].
// ---------------------------------------------------------------------------
__global__ void group_kernel(const float* __restrict__ xyz, const float* __restrict__ pts,
                             int Cpts, const float* __restrict__ nxyz,
                             const int* __restrict__ gi, int N, int S, int nsample,
                             int total, int Kpad, _Float16* __restrict__ X)
{
    int row = blockIdx.x * blockDim.x + threadIdx.x;
    if (row >= total) return;
    int c0 = row / nsample;           // b*S + s
    int b = c0 / S;
    int g = gi[row];
    const float* p = xyz + ((size_t)b * N + g) * 3;
    const float* q = nxyz + (size_t)c0 * 3;
    _Float16* xr = X + (size_t)row * Kpad;
    int k = 0;
    if (pts) {
        const float* pr = pts + ((size_t)b * N + g) * Cpts;
        for (int c = 0; c < Cpts; ++c) xr[k++] = (_Float16)pr[c];
    }
    xr[k++] = (_Float16)(p[0] - q[0]);
    xr[k++] = (_Float16)(p[1] - q[1]);
    xr[k++] = (_Float16)(p[2] - q[2]);
    for (; k < Kpad; ++k) xr[k] = (_Float16)0.f;
}

// ---------------------------------------------------------------------------
// Weights: f32 [K, C] -> f16 transposed [C, Kpad] (zero padded K).
// ---------------------------------------------------------------------------
__global__ void wt_cast(const float* __restrict__ W, int K, int C, int Kpad,
                        _Float16* __restrict__ Wt)
{
    int idx = blockIdx.x * blockDim.x + threadIdx.x;
    if (idx >= C * Kpad) return;
    int c = idx / Kpad, k = idx % Kpad;
    Wt[idx] = (k < K) ? (_Float16)W[(size_t)k * C + c] : (_Float16)0.f;
}

// ---------------------------------------------------------------------------
// TDM helper: issue tensor_load_to_lds of a [16 x chunk] f16 tile of
// Wt (row stride Kpad halfs) into LDS at ldsAddr.  2D descriptor,
// data_size = 2 bytes, tracked with TENSORcnt.
// ---------------------------------------------------------------------------
__device__ __forceinline__ void tdm_load_wtile(unsigned int ldsAddr,
                                               const _Float16* gsrc,
                                               int Kpad, int Cout, int chunk)
{
    unsigned long long ga = (unsigned long long)(uintptr_t)gsrc;
    v4u g0 = { 1u,                                   // count=1, no gather
               ldsAddr,                              // lds_addr
               (unsigned int)ga,                     // global_addr[31:0]
               (unsigned int)((ga >> 32) & 0x1FFFFFFu) | (2u << 30) }; // addr[56:32] | type=2
    v8i g1 = { (int)(1u << 16),                                   // data_size=1 (2 bytes)
               (int)(((unsigned)Kpad & 0xFFFFu) << 16),           // tensor_dim0 lo
               (int)((((unsigned)Kpad >> 16) & 0xFFFFu) |
                     (((unsigned)Cout & 0xFFFFu) << 16)),         // dim0 hi | tensor_dim1 lo
               (int)((((unsigned)Cout >> 16) & 0xFFFFu) |
                     (((unsigned)chunk & 0xFFFFu) << 16)),        // dim1 hi | tile_dim0
               (int)16,                                           // tile_dim1 = 16 rows
               (int)Kpad,                                         // tensor_dim0_stride lo32
               0, 0 };
    v4i gz = { 0, 0, 0, 0 };
#if __clang_major__ >= 23
    v8i gz8 = { 0, 0, 0, 0, 0, 0, 0, 0 };
    __builtin_amdgcn_tensor_load_to_lds(g0, g1, gz, gz, gz8, 0);
#else
    __builtin_amdgcn_tensor_load_to_lds(g0, g1, gz, gz, 0);
#endif
}

// ---------------------------------------------------------------------------
// WMMA GEMM: Y[M,Cout](f32) = X[M,Kpad](f16) @ Wt[Cout,Kpad](f16)^T
// Block = 4 waves, each wave owns a 64x16 output slab (4 accumulators) so one
// B-fragment feeds 4 WMMAs.  The block's 16-row weight tile is staged into
// LDS by the Tensor Data Mover in KCH-half chunks and consumed with ds loads;
// A streams from global (b128 pairs + prefetch).  Out-of-range M-subtiles are
// clamped (EXEC stays all-ones for WMMA), stores are guarded.
// ---------------------------------------------------------------------------
__global__ void __launch_bounds__(128) wmma_gemm(const _Float16* __restrict__ X,
                                                 const _Float16* __restrict__ Wt,
                                                 float* __restrict__ Y,
                                                 int Mpad, int Kpad, int Cout)
{
    __shared__ _Float16 sB[16 * KCH];
    int wave = threadIdx.x >> 5;
    int lane = threadIdx.x & 31;
    int mrow = lane & 15;
    int kb = (lane >> 4) * 8;
    int ntile = blockIdx.y;
    int mt0 = (blockIdx.x * 4 + wave) * 4;     // first of 4 m-subtiles

    int lastTile = Mpad / 16 - 1;
    int t0 = mt0 + 0 > lastTile ? lastTile : mt0 + 0;
    int t1 = mt0 + 1 > lastTile ? lastTile : mt0 + 1;
    int t2 = mt0 + 2 > lastTile ? lastTile : mt0 + 2;
    int t3 = mt0 + 3 > lastTile ? lastTile : mt0 + 3;
    const _Float16* a0p = X + (size_t)(t0 * 16 + mrow) * Kpad;
    const _Float16* a1p = X + (size_t)(t1 * 16 + mrow) * Kpad;
    const _Float16* a2p = X + (size_t)(t2 * 16 + mrow) * Kpad;
    const _Float16* a3p = X + (size_t)(t3 * 16 + mrow) * Kpad;

    v8f acc0 = {}, acc1 = {}, acc2 = {}, acc3 = {};

    for (int c0 = 0; c0 < Kpad; c0 += KCH) {
        int chunk = Kpad - c0;
        if (chunk > KCH) chunk = KCH;
        if (threadIdx.x < 32) {
            tdm_load_wtile((unsigned int)(uintptr_t)(&sB[0]),
                           Wt + (size_t)ntile * 16 * Kpad + c0, Kpad, Cout, chunk);
            __builtin_amdgcn_s_wait_tensorcnt(0);
        }
        __syncthreads();
        for (int kk = 0; kk < chunk; kk += 32) {
            int k0 = c0 + kk;
            const _Float16* bL = sB + (size_t)mrow * chunk + kk;
            v8h b0 = *(const v8h*)(bL + kb);            // ds_load_b128
            v8h b1 = *(const v8h*)(bL + kb + 16);       // ds_load_b128
            v16h bb = __builtin_shufflevector(b0, b1, 0, 1, 2, 3, 4, 5, 6, 7,
                                              8, 9, 10, 11, 12, 13, 14, 15);
            __builtin_prefetch(a0p + k0 + 512, 0, 1);   // global_prefetch_b8
            v8h x0 = *(const v8h*)(a0p + k0 + kb);
            v8h x1 = *(const v8h*)(a0p + k0 + kb + 16);
            v8h x2 = *(const v8h*)(a1p + k0 + kb);
            v8h x3 = *(const v8h*)(a1p + k0 + kb + 16);
            v8h x4 = *(const v8h*)(a2p + k0 + kb);
            v8h x5 = *(const v8h*)(a2p + k0 + kb + 16);
            v8h x6 = *(const v8h*)(a3p + k0 + kb);
            v8h x7 = *(const v8h*)(a3p + k0 + kb + 16);
            v16h a0 = __builtin_shufflevector(x0, x1, 0, 1, 2, 3, 4, 5, 6, 7,
                                              8, 9, 10, 11, 12, 13, 14, 15);
            v16h a1 = __builtin_shufflevector(x2, x3, 0, 1, 2, 3, 4, 5, 6, 7,
                                              8, 9, 10, 11, 12, 13, 14, 15);
            v16h a2 = __builtin_shufflevector(x4, x5, 0, 1, 2, 3, 4, 5, 6, 7,
                                              8, 9, 10, 11, 12, 13, 14, 15);
            v16h a3 = __builtin_shufflevector(x6, x7, 0, 1, 2, 3, 4, 5, 6, 7,
                                              8, 9, 10, 11, 12, 13, 14, 15);
            acc0 = __builtin_amdgcn_wmma_f32_16x16x32_f16(false, a0, false, bb, (short)0, acc0, false, false);
            acc1 = __builtin_amdgcn_wmma_f32_16x16x32_f16(false, a1, false, bb, (short)0, acc1, false, false);
            acc2 = __builtin_amdgcn_wmma_f32_16x16x32_f16(false, a2, false, bb, (short)0, acc2, false, false);
            acc3 = __builtin_amdgcn_wmma_f32_16x16x32_f16(false, a3, false, bb, (short)0, acc3, false, false);
        }
        __syncthreads();
    }

    int n = ntile * 16 + mrow;
    int hi = (lane >> 4) ? 8 : 0;
    if ((mt0 + 0) * 16 < Mpad) {
        int mb = t0 * 16 + hi;
        for (int r = 0; r < 8; ++r) Y[(size_t)(mb + r) * Cout + n] = acc0[r];
    }
    if ((mt0 + 1) * 16 < Mpad) {
        int mb = t1 * 16 + hi;
        for (int r = 0; r < 8; ++r) Y[(size_t)(mb + r) * Cout + n] = acc1[r];
    }
    if ((mt0 + 2) * 16 < Mpad) {
        int mb = t2 * 16 + hi;
        for (int r = 0; r < 8; ++r) Y[(size_t)(mb + r) * Cout + n] = acc2[r];
    }
    if ((mt0 + 3) * 16 < Mpad) {
        int mb = t3 * 16 + hi;
        for (int r = 0; r < 8; ++r) Y[(size_t)(mb + r) * Cout + n] = acc3[r];
    }
}

// ---------------------------------------------------------------------------
// BatchNorm statistics over M rows (one block per channel) + bn_relu + casts.
// ---------------------------------------------------------------------------
__global__ void bn_stats(const float* __restrict__ Y, int M, int C,
                         float* __restrict__ mv)
{
    int c = blockIdx.x;
    float s = 0.f, s2 = 0.f;
    for (int m = threadIdx.x; m < M; m += blockDim.x) {
        float v = Y[(size_t)m * C + c];
        s += v; s2 += v * v;
    }
    __shared__ float ss[256], sq[256];
    ss[threadIdx.x] = s; sq[threadIdx.x] = s2;
    __syncthreads();
    for (int o = 128; o > 0; o >>= 1) {
        if ((int)threadIdx.x < o) {
            ss[threadIdx.x] += ss[threadIdx.x + o];
            sq[threadIdx.x] += sq[threadIdx.x + o];
        }
        __syncthreads();
    }
    if (threadIdx.x == 0) {
        float mean = ss[0] / (float)M;
        float var = sq[0] / (float)M - mean * mean;
        mv[2 * c] = mean; mv[2 * c + 1] = var;
    }
}

__global__ void bn_relu(float* __restrict__ Y, const float* __restrict__ mv,
                        const float* __restrict__ g, const float* __restrict__ b,
                        size_t total, int C)
{
    size_t idx = (size_t)blockIdx.x * blockDim.x + threadIdx.x;
    if (idx >= total) return;
    int c = (int)(idx % C);
    float mean = mv[2 * c], var = mv[2 * c + 1];
    float v = (Y[idx] - mean) * rsqrtf(var + BN_EPS) * g[c] + b[c];
    Y[idx] = v > 0.f ? v : 0.f;
}

__global__ void cast_pad(const float* __restrict__ Y, int Mreal, int C, int Kpad,
                         _Float16* __restrict__ X, int Mpad)
{
    size_t idx = (size_t)blockIdx.x * blockDim.x + threadIdx.x;
    if (idx >= (size_t)Mpad * Kpad) return;
    int m = (int)(idx / Kpad), k = (int)(idx % Kpad);
    X[idx] = (m < Mreal && k < C) ? (_Float16)Y[(size_t)m * C + k] : (_Float16)0.f;
}

__global__ void copy_cast(const float* __restrict__ src, int M, int C, int Kpad,
                          int dstOff, _Float16* __restrict__ X)
{
    size_t idx = (size_t)blockIdx.x * blockDim.x + threadIdx.x;
    if (idx >= (size_t)M * C) return;
    int m = (int)(idx / C), c = (int)(idx % C);
    X[(size_t)m * Kpad + dstOff + c] = (_Float16)src[idx];
}

__global__ void copy_f32(const float* __restrict__ src, float* __restrict__ dst,
                         size_t total)
{
    size_t idx = (size_t)blockIdx.x * blockDim.x + threadIdx.x;
    if (idx < total) dst[idx] = src[idx];
}

__global__ void maxpool(const float* __restrict__ Y, int R, int nsample, int C,
                        float* __restrict__ out, int outStride, int outOff)
{
    size_t idx = (size_t)blockIdx.x * blockDim.x + threadIdx.x;
    if (idx >= (size_t)R * C) return;
    int r = (int)(idx / C), c = (int)(idx % C);
    const float* p = Y + (size_t)r * nsample * C + c;
    float m = -1e30f;
    for (int j = 0; j < nsample; ++j) {
        float v = p[(size_t)j * C];
        m = v > m ? v : m;
    }
    out[(size_t)r * outStride + outOff + c] = m;
}

// ---------------------------------------------------------------------------
// Feature propagation: inverse-distance 3-NN interpolation -> f16 into X.
// ---------------------------------------------------------------------------
__global__ void three_interp(const float* __restrict__ xyz1, const float* __restrict__ xyz2,
                             const float* __restrict__ pts2, int N, int S, int C,
                             int total, int Kpad, int dstOff, _Float16* __restrict__ X)
{
    int t = blockIdx.x * blockDim.x + threadIdx.x;
    if (t >= total) return;
    int b = t / N;
    const float* p = xyz1 + (size_t)t * 3;
    float px = p[0], py = p[1], pz = p[2];
    const float* src = xyz2 + (size_t)b * S * 3;
    float d0 = 1e30f, d1 = 1e30f, d2 = 1e30f;
    int i0 = 0, i1 = 0, i2 = 0;
    for (int s = 0; s < S; ++s) {
        float dx = src[s * 3 + 0] - px;
        float dy = src[s * 3 + 1] - py;
        float dz = src[s * 3 + 2] - pz;
        float d = dx * dx + dy * dy + dz * dz;
        if (d < d0)      { d2 = d1; i2 = i1; d1 = d0; i1 = i0; d0 = d; i0 = s; }
        else if (d < d1) { d2 = d1; i2 = i1; d1 = d;  i1 = s; }
        else if (d < d2) { d2 = d;  i2 = s; }
    }
    float w0 = 1.f / (d0 + 1e-8f), w1 = 1.f / (d1 + 1e-8f), w2 = 1.f / (d2 + 1e-8f);
    float ws = w0 + w1 + w2;
    w0 /= ws; w1 /= ws; w2 /= ws;
    const float* r0 = pts2 + ((size_t)b * S + i0) * C;
    const float* r1 = pts2 + ((size_t)b * S + i1) * C;
    const float* r2 = pts2 + ((size_t)b * S + i2) * C;
    _Float16* xr = X + (size_t)t * Kpad + dstOff;
    for (int c = 0; c < C; ++c)
        xr[c] = (_Float16)(w0 * r0[c] + w1 * r1[c] + w2 * r2[c]);
}

__global__ void bcast_interp(const float* __restrict__ pts2, int Nn, int C,
                             size_t total, int Kpad, int dstOff,
                             _Float16* __restrict__ X)
{
    size_t idx = (size_t)blockIdx.x * blockDim.x + threadIdx.x;
    if (idx >= total) return;
    int c = (int)(idx % C);
    size_t row = idx / C;
    int b = (int)(row / Nn);
    X[row * Kpad + dstOff + c] = (_Float16)pts2[(size_t)b * C + c];
}

// ---------------------------------------------------------------------------
// Host orchestration
// ---------------------------------------------------------------------------
extern "C" void kernel_launch(void* const* d_in, const int* in_sizes, int n_in,
                              void* d_out, int out_size, void* d_ws, size_t ws_size,
                              hipStream_t stream)
{
    (void)in_sizes; (void)n_in; (void)out_size; (void)ws_size;
    const int B = 8, N0 = 8192;
    int cur = 0;
    const float* xyz = (const float*)d_in[cur++];

    struct Layer { const float *W, *g, *b; int Cin, Cout; };
    auto take = [&](int cin, int cout) {
        Layer L;
        L.W = (const float*)d_in[cur];
        L.g = (const float*)d_in[cur + 1];
        L.b = (const float*)d_in[cur + 2];
        L.Cin = cin; L.Cout = cout;
        cur += 3;
        return L;
    };

    static const int sa1_dims[3][4] = {{3,32,32,64},{3,64,64,128},{3,64,96,128}};
    static const int sa2_dims[3][4] = {{323,64,64,128},{323,128,128,256},{323,128,128,256}};
    static const int sa3_dims[3][4] = {{643,256,256,512},{643,256,512,512},{643,256,512,512}};
    Layer sa1L[3][3], sa2L[3][3], sa3L[3][3];
    for (int s = 0; s < 3; ++s)
        for (int l = 0; l < 3; ++l)
            sa1L[s][l] = take(sa1_dims[s][l], sa1_dims[s][l + 1]);
    for (int s = 0; s < 3; ++s)
        for (int l = 0; l < 3; ++l)
            sa2L[s][l] = take(sa2_dims[s][l], sa2_dims[s][l + 1]);
    for (int s = 0; s < 3; ++s)
        for (int l = 0; l < 3; ++l)
            sa3L[s][l] = take(sa3_dims[s][l], sa3_dims[s][l + 1]);
    Layer fp3L[2] = { take(2176, 1024), take(1024, 512) };
    Layer fp2L[2] = { take(832, 512),   take(512, 512) };
    Layer fp1L[2] = { take(512, 256),   take(256, 128) };
    Layer pgL = take(1536, 256);
    Layer plL = take(128, 256);

    // ---- workspace bump allocator ----
    char* base = (char*)d_ws;
    size_t off = 0;
    auto alloc = [&](size_t bytes) -> void* {
        void* p = base + off;
        off += (bytes + 255) & ~(size_t)255;
        return p;
    };
    const size_t XCAP = 34ull * 1024 * 1024;  // halfs  (max: fp1 65536x512)
    const size_t YCAP = 34ull * 1024 * 1024;  // floats (max: sa1-s3 262144x128)
    _Float16* X  = (_Float16*)alloc(XCAP * sizeof(_Float16));
    float*    Y  = (float*)alloc(YCAP * sizeof(float));
    _Float16* Wt = (_Float16*)alloc(4ull * 1024 * 1024 * sizeof(_Float16));
    float* mv    = (float*)alloc(2 * 1024 * sizeof(float));
    float* dist  = (float*)alloc((size_t)B * N0 * sizeof(float));
    int* fpsIdx  = (int*)alloc((size_t)B * 512 * sizeof(int));
    int* gi      = (int*)alloc((size_t)B * 512 * 64 * sizeof(int));
    float* l1x = (float*)alloc((size_t)B * 512 * 3 * sizeof(float));
    float* l1p = (float*)alloc((size_t)B * 512 * 320 * sizeof(float));
    float* l2x = (float*)alloc((size_t)B * 128 * 3 * sizeof(float));
    float* l2p = (float*)alloc((size_t)B * 128 * 640 * sizeof(float));
    float* l3x = (float*)alloc((size_t)B * 3 * sizeof(float));
    float* l3p = (float*)alloc((size_t)B * 1536 * sizeof(float));
    float* f3o = (float*)alloc((size_t)B * 128 * 512 * sizeof(float));
    float* f2o = (float*)alloc((size_t)B * 512 * 512 * sizeof(float));
    float* l0p = (float*)alloc((size_t)B * 8192 * 128 * sizeof(float));

    // ---- MLP chain (X f16 in-place ping; Y f32 scratch) ----
    auto run_mlp = [&](int Mreal, int Mpad, const Layer* Ls, int nL,
                       int mode /*0=maxpool, 1=copy rows*/, float* dst,
                       int R, int nsample, int outStride, int outOff) {
        for (int i = 0; i < nL; ++i) {
            const Layer& L = Ls[i];
            int KpadL = rup(L.Cin, 32);
            int wtN = L.Cout * KpadL;
            wt_cast<<<(wtN + 255) / 256, 256, 0, stream>>>(L.W, L.Cin, L.Cout, KpadL, Wt);
            dim3 gg((unsigned)((Mpad + 255) / 256), (unsigned)(L.Cout / 16));
            wmma_gemm<<<gg, 128, 0, stream>>>(X, Wt, Y, Mpad, KpadL, L.Cout);
            bn_stats<<<L.Cout, 256, 0, stream>>>(Y, Mreal, L.Cout, mv);
            size_t tot = (size_t)Mreal * L.Cout;
            bn_relu<<<(unsigned)((tot + 255) / 256), 256, 0, stream>>>(Y, mv, L.g, L.b, tot, L.Cout);
            if (i + 1 < nL) {
                int Kn = rup(L.Cout, 32);
                size_t t2 = (size_t)Mpad * Kn;
                cast_pad<<<(unsigned)((t2 + 255) / 256), 256, 0, stream>>>(Y, Mreal, L.Cout, Kn, X, Mpad);
            } else if (mode == 0) {
                size_t t2 = (size_t)R * L.Cout;
                maxpool<<<(unsigned)((t2 + 255) / 256), 256, 0, stream>>>(Y, R, nsample, L.Cout, dst, outStride, outOff);
            } else {
                size_t t2 = (size_t)Mreal * L.Cout;
                copy_f32<<<(unsigned)((t2 + 255) / 256), 256, 0, stream>>>(Y, dst, t2);
            }
        }
    };

    auto run_sa_scale = [&](const float* sxyz, int Nsrc, const float* spts, int Cpts,
                            const float* nx, int S, float radius, int nsample,
                            const Layer* Ls, float* outFeat, int outStride, int outOff) {
        int totalC = B * S;
        ball_query<<<(totalC + 7) / 8, 256, 0, stream>>>(sxyz, nx, Nsrc, S, totalC,
                                                         radius * radius, nsample, gi);
        int M = B * S * nsample;
        int Kpad0 = rup(Cpts + 3, 32);
        group_kernel<<<(M + 255) / 256, 256, 0, stream>>>(sxyz, spts, Cpts, nx, gi,
                                                          Nsrc, S, nsample, M, Kpad0, X);
        run_mlp(M, M, Ls, 3, 0, outFeat, B * S, nsample, outStride, outOff);
    };

    // ---- SA1 ----
    fps_kernel<<<B, 512, 0, stream>>>(xyz, dist, N0, 512, fpsIdx);
    gather_xyz<<<(B * 512 + 255) / 256, 256, 0, stream>>>(xyz, fpsIdx, N0, 512, B * 512, l1x);
    run_sa_scale(xyz, N0, nullptr, 0, l1x, 512, 0.05f, 16, sa1L[0], l1p, 320, 0);
    run_sa_scale(xyz, N0, nullptr, 0, l1x, 512, 0.10f, 32, sa1L[1], l1p, 320, 64);
    run_sa_scale(xyz, N0, nullptr, 0, l1x, 512, 0.20f, 64, sa1L[2], l1p, 320, 192);

    // ---- SA2 ----
    fps_kernel<<<B, 512, 0, stream>>>(l1x, dist, 512, 128, fpsIdx);
    gather_xyz<<<(B * 128 + 255) / 256, 256, 0, stream>>>(l1x, fpsIdx, 512, 128, B * 128, l2x);
    run_sa_scale(l1x, 512, l1p, 320, l2x, 128, 0.10f, 16, sa2L[0], l2p, 640, 0);
    run_sa_scale(l1x, 512, l1p, 320, l2x, 128, 0.20f, 32, sa2L[1], l2p, 640, 128);
    run_sa_scale(l1x, 512, l1p, 320, l2x, 128, 0.40f, 64, sa2L[2], l2p, 640, 384);

    // ---- SA3 (npoint = 1) ----
    fps_kernel<<<B, 512, 0, stream>>>(l2x, dist, 128, 1, fpsIdx);
    gather_xyz<<<1, 256, 0, stream>>>(l2x, fpsIdx, 128, 1, B, l3x);
    run_sa_scale(l2x, 128, l2p, 640, l3x, 1, 0.40f, 32,  sa3L[0], l3p, 1536, 0);
    run_sa_scale(l2x, 128, l2p, 640, l3x, 1, 0.80f, 64,  sa3L[1], l3p, 1536, 512);
    run_sa_scale(l2x, 128, l2p, 640, l3x, 1, 1.20f, 128, sa3L[2], l3p, 1536, 1024);

    // ---- FP3: l2p(640) ++ broadcast(l3p,1536) -> [1024,512] ----
    {
        int M = B * 128, Kpad = 2176;
        size_t t1 = (size_t)M * 640;
        copy_cast<<<(unsigned)((t1 + 255) / 256), 256, 0, stream>>>(l2p, M, 640, Kpad, 0, X);
        size_t t2 = (size_t)M * 1536;
        bcast_interp<<<(unsigned)((t2 + 255) / 256), 256, 0, stream>>>(l3p, 128, 1536, t2, Kpad, 640, X);
        run_mlp(M, M, fp3L, 2, 1, f3o, 0, 0, 0, 0);
    }
    // ---- FP2: l1p(320) ++ interp(f3o,512) -> [512,512] ----
    {
        int M = B * 512, Kpad = 832;
        size_t t1 = (size_t)M * 320;
        copy_cast<<<(unsigned)((t1 + 255) / 256), 256, 0, stream>>>(l1p, M, 320, Kpad, 0, X);
        three_interp<<<(M + 255) / 256, 256, 0, stream>>>(l1x, l2x, f3o, 512, 128, 512, M, Kpad, 320, X);
        run_mlp(M, M, fp2L, 2, 1, f2o, 0, 0, 0, 0);
    }
    // ---- FP1: interp(f2o,512) -> [256,128] ----
    {
        int M = B * N0, Kpad = 512;
        three_interp<<<(M + 255) / 256, 256, 0, stream>>>(xyz, l1x, f2o, N0, 512, 512, M, Kpad, 0, X);
        run_mlp(M, M, fp1L, 2, 1, l0p, 0, 0, 0, 0);
    }
    // ---- proj_global: l3p [8,1536] -> 256, max over S=1 -> d_out[:, 0:256] ----
    {
        size_t t = (size_t)16 * 1536;
        cast_pad<<<(unsigned)((t + 255) / 256), 256, 0, stream>>>(l3p, B, 1536, 1536, X, 16);
        run_mlp(B, 16, &pgL, 1, 0, (float*)d_out, B, 1, 512, 0);
    }
    // ---- proj_local: l0p [65536,128] -> 256, max over N -> d_out[:, 256:512] ----
    {
        int M = B * N0;
        size_t t = (size_t)M * 128;
        cast_pad<<<(unsigned)((t + 255) / 256), 256, 0, stream>>>(l0p, M, 128, 128, X, M);
        run_mlp(M, M, &plL, 1, 0, (float*)d_out, B, N0, 512, 256);
    }
}
